// ChargeModel_5609227288972
// MI455X (gfx1250) — compile-verified, matrix operand order checked by reading, not run on hardware
//
#include <hip/hip_runtime.h>

typedef __bf16 bf16_t;
typedef __attribute__((ext_vector_type(16))) __bf16 v16bf;
typedef __attribute__((ext_vector_type(8)))  float  v8f;

#define NN 128   // nodes per graph
#define HH 64    // hidden width
#define EE 1024  // edges per graph

// ---------------------------------------------------------------------------
// Prep: convert W1,W2,W3 (fp32, row-major [K][64]) into bf16 transposed
// [64][K] so each WMMA B-fragment lane reads contiguous K pairs.
// Layout in ws: W1t (64*128) | W2t (64*64) | W3t (64*64)
// ---------------------------------------------------------------------------
__global__ void prep_weights(const float* __restrict__ W1,
                             const float* __restrict__ W2,
                             const float* __restrict__ W3,
                             bf16_t* __restrict__ wt) {
    int idx = blockIdx.x * 256 + threadIdx.x;
    if (idx < 64 * 128) {
        int n = idx >> 7, k = idx & 127;
        wt[idx] = (bf16_t)W1[k * 64 + n];
    } else if (idx < 64 * 128 + 64 * 64) {
        int j = idx - 64 * 128;
        int n = j >> 6, k = j & 63;
        wt[idx] = (bf16_t)W2[k * 64 + n];
    } else if (idx < 64 * 128 + 2 * 64 * 64) {
        int j = idx - (64 * 128 + 64 * 64);
        int n = j >> 6, k = j & 63;
        wt[idx] = (bf16_t)W3[k * 64 + n];
    }
}

// ---------------------------------------------------------------------------
// GEMM: H[128][64] (fp32, LDS) = A[128][K] x Wt^T, bf16 inputs, fp32 acc.
// 8 waves: wave w owns rows [16w,16w+16), 4 N-tiles of 16.
// CDNA5 16-bit A layout: lane = M (0..15, both halves), VGPR v / half-wave
// select K pairs: k = (v>>2)*16 + half*8 + (v&3)*2.  B mirrors with lane = N.
// ---------------------------------------------------------------------------
template <int K, bool AGLOBAL>
__device__ __forceinline__ void gemm_bf16(const float* __restrict__ Ag,
                                          const bf16_t* __restrict__ Al,
                                          const bf16_t* __restrict__ Wt,
                                          float* __restrict__ Hout,
                                          int wave, int lane) {
    const int m0   = wave << 4;
    const int half = lane >> 4;
    const int lm   = lane & 15;

    v8f acc[4] = {};  // 4 N-tiles of 16x16 fp32

#pragma unroll
    for (int kk = 0; kk < K; kk += 32) {
        // ---- A fragment (row m0+lm, K pairs per layout) ----
        v16bf a;
        if constexpr (AGLOBAL) {
            const float* xr = Ag + (m0 + lm) * K;
#pragma unroll
            for (int v = 0; v < 8; ++v) {
                int k = kk + ((v >> 2) << 4) + (half << 3) + ((v & 3) << 1);
                a[2 * v]     = (bf16_t)xr[k];
                a[2 * v + 1] = (bf16_t)xr[k + 1];
            }
        } else {
            const bf16_t* ar = Al + (m0 + lm) * K;
#pragma unroll
            for (int v = 0; v < 8; ++v) {
                int k = kk + ((v >> 2) << 4) + (half << 3) + ((v & 3) << 1);
                a[2 * v]     = ar[k];
                a[2 * v + 1] = ar[k + 1];
            }
        }
        // ---- B fragments + WMMA for each N tile ----
#pragma unroll
        for (int t = 0; t < 4; ++t) {
            v16bf b;
            const bf16_t* wr = Wt + (t * 16 + lm) * K;  // Wt is [N][K]
#pragma unroll
            for (int v = 0; v < 8; ++v) {
                int k = kk + ((v >> 2) << 4) + (half << 3) + ((v & 3) << 1);
                b[2 * v]     = wr[k];
                b[2 * v + 1] = wr[k + 1];
            }
            acc[t] = __builtin_amdgcn_wmma_f32_16x16x32_bf16(
                false, a, false, b, (short)0, acc[t], false, false);
        }
    }

    // ---- store C: VGPR r holds M = r (lanes 0-15) / r+8 (lanes 16-31) ----
#pragma unroll
    for (int t = 0; t < 4; ++t)
#pragma unroll
        for (int r = 0; r < 8; ++r)
            Hout[(m0 + r + 8 * half) * HH + t * 16 + lm] = acc[t][r];
}

// ---------------------------------------------------------------------------
// Symmetric-norm scatter-add aggregation:
//   agg[i][:]  = dinv[i]^2 * h[i][:]                      (self loop)
//   agg[c][:] += dinv[r]*dinv[c] * h[r][:]  for each edge (r->c)
// 64 threads cooperate per edge (one channel each): conflict-free LDS banks.
// ---------------------------------------------------------------------------
__device__ __forceinline__ void aggregate(const int* __restrict__ rowE,
                                          const int* __restrict__ colE,
                                          const float* __restrict__ h,
                                          float* __restrict__ agg,
                                          const float* __restrict__ dinv,
                                          int tid) {
    for (int idx = tid; idx < NN * HH; idx += 256) {
        int i = idx >> 6;
        float d = dinv[i];
        agg[idx] = d * d * h[idx];
    }
    __syncthreads();
    const int ch = tid & 63;
    const int eg = tid >> 6;  // 4 edges in flight
    for (int e = eg; e < EE; e += 4) {
        int r = rowE[e], c = colE[e];
        float nrm = dinv[r] * dinv[c];
        atomicAdd(&agg[c * HH + ch], nrm * h[r * HH + ch]);
    }
    __syncthreads();
}

__device__ __forceinline__ void sig_bias_to_bf16(const float* __restrict__ agg,
                                                 const float* __restrict__ bias,
                                                 bf16_t* __restrict__ a_s,
                                                 int tid) {
    for (int idx = tid; idx < NN * HH; idx += 256) {
        float v = agg[idx] + bias[idx & 63];
        a_s[idx] = (bf16_t)(1.0f / (1.0f + __expf(-v)));
    }
}

// ---------------------------------------------------------------------------
// Main kernel: one workgroup (8 wave32) per graph.
// ---------------------------------------------------------------------------
__global__ __launch_bounds__(256) void gcn3_kernel(
    const float* __restrict__ x, const int* __restrict__ ei,
    const bf16_t* __restrict__ wt, const float* __restrict__ b1,
    const float* __restrict__ b2, const float* __restrict__ b3,
    float* __restrict__ out) {
    __shared__ float  h_s[NN * HH];    // 32 KB  GEMM output
    __shared__ float  agg_s[NN * HH];  // 32 KB  aggregation (fp32 atomics)
    __shared__ bf16_t a_s[NN * HH];    // 16 KB  next-layer A matrix (bf16)
    __shared__ float  dinv_s[NN];
    __shared__ int    deg_s[NN];
    __shared__ float  red_s;

    const int g    = blockIdx.x;
    const int tid  = threadIdx.x;
    const int wave = tid >> 5;
    const int lane = tid & 31;

    const int*   rowE = ei + (size_t)g * 2 * EE;  // sources
    const int*   colE = rowE + EE;                // targets
    const float* xg   = x + (size_t)g * NN * NN;

    // ---- degrees (shared by all 3 layers): deg[i] = 1 + #incoming ----
    if (tid < NN) deg_s[tid] = 1;
    __syncthreads();
    for (int e = tid; e < EE; e += 256) atomicAdd(&deg_s[colE[e]], 1);
    __syncthreads();
    if (tid < NN) dinv_s[tid] = rsqrtf((float)deg_s[tid]);
    __syncthreads();

    // ---- layer 1: h = x @ W1 ; aggregate ; sigmoid(+b1) ----
    gemm_bf16<128, true>(xg, nullptr, wt, h_s, wave, lane);
    __syncthreads();
    aggregate(rowE, colE, h_s, agg_s, dinv_s, tid);
    sig_bias_to_bf16(agg_s, b1, a_s, tid);
    __syncthreads();

    // ---- layer 2 ----
    gemm_bf16<64, false>(nullptr, a_s, wt + 64 * 128, h_s, wave, lane);
    __syncthreads();
    aggregate(rowE, colE, h_s, agg_s, dinv_s, tid);
    sig_bias_to_bf16(agg_s, b2, a_s, tid);
    __syncthreads();

    // ---- layer 3 (no sigmoid) ----
    gemm_bf16<64, false>(nullptr, a_s, wt + 64 * 128 + 64 * 64, h_s, wave, lane);
    __syncthreads();
    aggregate(rowE, colE, h_s, agg_s, dinv_s, tid);

    // ---- mean over nodes & channels; bias folds to 128 * sum(b3) ----
    float part = 0.f;
    for (int idx = tid; idx < NN * HH; idx += 256) part += agg_s[idx];
    if (tid == 0) red_s = 0.f;
    __syncthreads();
    atomicAdd(&red_s, part);
    __syncthreads();
    if (tid == 0) {
        float bs = 0.f;
        for (int c = 0; c < HH; ++c) bs += b3[c];
        out[g] = (red_s + (float)NN * bs) * (1.0f / (float)(NN * HH));
    }
}

// ---------------------------------------------------------------------------
extern "C" void kernel_launch(void* const* d_in, const int* in_sizes, int n_in,
                              void* d_out, int out_size, void* d_ws,
                              size_t ws_size, hipStream_t stream) {
    const float* x  = (const float*)d_in[0];
    const int*   ei = (const int*)d_in[1];
    const float* W1 = (const float*)d_in[2];
    const float* b1 = (const float*)d_in[3];
    const float* W2 = (const float*)d_in[4];
    const float* b2 = (const float*)d_in[5];
    const float* W3 = (const float*)d_in[6];
    const float* b3 = (const float*)d_in[7];
    float*       out = (float*)d_out;
    bf16_t*      wt  = (bf16_t*)d_ws;

    const int B = in_sizes[0] / (NN * NN);

    prep_weights<<<64, 256, 0, stream>>>(W1, W2, W3, wt);
    gcn3_kernel<<<B, 256, 0, stream>>>(x, ei, wt, b1, b2, b3, out);
}
